// ImageTransformerDenoiserModelV2_44143673868629
// MI455X (gfx1250) — compile-verified, matrix operand order checked by reading, not run on hardware
//
#include <hip/hip_runtime.h>
#include <math.h>

// ---------------- types ----------------
typedef __attribute__((ext_vector_type(16))) __bf16        v16bf;
typedef __attribute__((ext_vector_type(8)))  float         v8f;
typedef __attribute__((ext_vector_type(4)))  unsigned int  u4_t;
typedef __attribute__((ext_vector_type(4)))  float         f4_t;

// ---------------- problem constants ----------------
constexpr int BB = 2, HH = 64, WW = 64;
constexpr int DM = 256, DH = 64, NHEADS = 4, DFF = 768, CD = 256, KW = 7;
constexpr int NPIX = BB * HH * WW;              // 8192 tokens
constexpr float EPSV = 1e-6f;

#define DEV static __device__ __forceinline__

DEV unsigned short f2bf(float f) {              // round-to-nearest-even
  unsigned int u = __float_as_uint(f);
  u += 0x7FFFu + ((u >> 16) & 1u);
  return (unsigned short)(u >> 16);
}
DEV float bf2f(unsigned short s) { return __uint_as_float(((unsigned int)s) << 16); }

// ---------------- ada scale: out[b,d] = cond[b,:]·w[d,:] + 1 ----------------
__global__ void ada_scale_kernel(const float* __restrict__ cond,
                                 const float* __restrict__ w,
                                 float* __restrict__ out) {
  int idx = blockIdx.x * blockDim.x + threadIdx.x;     // over BB*DM
  if (idx >= BB * DM) return;
  int b = idx / DM, d = idx % DM;
  const float* cr = cond + b * CD;
  const float* wr = w + (size_t)d * CD;
  float s = 0.f;
  #pragma unroll 4
  for (int c = 0; c < CD; ++c) s += cr[c] * wr[c];
  out[idx] = s + 1.0f;
}

// ---------------- f32 -> bf16 array cast ----------------
__global__ void cast_bf16_kernel(const float* __restrict__ in,
                                 unsigned short* __restrict__ out, int n) {
  int i = blockIdx.x * blockDim.x + threadIdx.x;
  if (i < n) out[i] = f2bf(in[i]);
}

// ---------------- adaRMS norm, one wave per 256-elem row, bf16 out ----------
__global__ void __launch_bounds__(256)
rms_to_bf16_kernel(const float* __restrict__ x, const float* __restrict__ ada,
                   unsigned short* __restrict__ xn, int rows_per_batch) {
  int row  = blockIdx.x * 8 + (threadIdx.x >> 5);
  int lane = threadIdx.x & 31;
  const float* xr = x + (size_t)row * DM + lane * 8;
  f4_t a = *(const f4_t*)(xr);
  f4_t b = *(const f4_t*)(xr + 4);
  float ss = a.x*a.x + a.y*a.y + a.z*a.z + a.w*a.w
           + b.x*b.x + b.y*b.y + b.z*b.z + b.w*b.w;
  #pragma unroll
  for (int off = 16; off; off >>= 1) ss += __shfl_xor(ss, off, 32);
  float inv = rsqrtf(ss * (1.0f / DM) + EPSV);
  int bidx = row / rows_per_batch;
  const float* ar = ada + bidx * DM + lane * 8;
  unsigned short* op = xn + (size_t)row * DM + lane * 8;
  float vals[8] = {a.x, a.y, a.z, a.w, b.x, b.y, b.z, b.w};
  #pragma unroll
  for (int i = 0; i < 8; ++i) op[i] = f2bf(vals[i] * ar[i] * inv);
}

// ---------------- WMMA GEMM: C[M,N] = A[M,K]·Bt[N,K]^T (+skip) ---------------
// A, Bt bf16 row-major; wave -> 16x64 strip; 8 waves -> 128x64 block tile.
template <bool ADD_SKIP, bool OUT_BF16>
__global__ void __launch_bounds__(256)
gemm_bf16_wmma(const unsigned short* __restrict__ A,
               const unsigned short* __restrict__ Bt,
               const float* __restrict__ skip,
               void* __restrict__ Cout, int M, int N, int Kd) {
  const int lane = threadIdx.x & 31;
  const int wv   = threadIdx.x >> 5;
  const int half = lane >> 4;          // K-octet selector per ISA bf16 layout
  const int lr   = lane & 15;          // row (A) / col (B) within 16
  const int m0   = blockIdx.y * 128 + wv * 16;
  const int n0   = blockIdx.x * 64;

  v8f acc[4] = {};
  const unsigned short* Arow = A + (size_t)(m0 + lr) * Kd;

  for (int k0 = 0; k0 < Kd; k0 += 32) {
    union { u4_t u[2]; v16bf v; } afrag;
    const u4_t* pa = (const u4_t*)(Arow + k0 + 8 * half);
    afrag.u[0] = pa[0];                // K = k0+8h .. k0+8h+7
    afrag.u[1] = pa[2];                // K = k0+16+8h .. k0+16+8h+7
    #pragma unroll
    for (int t = 0; t < 4; ++t) {
      union { u4_t u[2]; v16bf v; } bfrag;
      const u4_t* pb = (const u4_t*)(Bt + (size_t)(n0 + t * 16 + lr) * Kd + k0 + 8 * half);
      bfrag.u[0] = pb[0];
      bfrag.u[1] = pb[2];
      acc[t] = __builtin_amdgcn_wmma_f32_16x16x32_bf16(
          false, afrag.v, false, bfrag.v, (short)0, acc[t], false, false);
    }
  }

  // C/D layout: vgpr i -> row m0 + 8*half + i, col n0 + 16t + lr
  #pragma unroll
  for (int t = 0; t < 4; ++t) {
    int c = n0 + t * 16 + lr;
    #pragma unroll
    for (int i = 0; i < 8; ++i) {
      int r = m0 + half * 8 + i;
      float v = acc[t][i];
      if (ADD_SKIP) v += skip[(size_t)r * N + c];
      if (OUT_BF16) ((unsigned short*)Cout)[(size_t)r * N + c] = f2bf(v);
      else          ((float*)Cout)[(size_t)r * N + c] = v;
    }
  }
}

// ---------------- axial RoPE (first 32 dims) + q scale, in place ------------
__global__ void __launch_bounds__(256)
rope_kernel(float* __restrict__ qkv, const float* __restrict__ pos) {
  int gid  = blockIdx.x * blockDim.x + threadIdx.x;
  int t    = gid & 31;                 // 32 threads per (pixel, head)
  int ph   = gid >> 5;
  int head = ph & (NHEADS - 1);
  int p    = ph >> 2;
  int h = (p >> 6) & 63, w = p & 63;
  float* q = qkv + (size_t)p * (3 * DM) + head * DH;
  float* k = q + DM;
  const float QS = 0.125f;             // 1/sqrt(D_HEAD)
  if (t < 16) {
    float pv = pos[(h * WW + w) * 2 + (t < 8 ? 0 : 1)];
    // freqs_j = pi * 10^(j/8), j = t & 7
    float fr = 3.14159265358979f * exp2f((float)(t & 7) * 0.41524101186f);
    float f  = pv * fr;
    float c = cosf(f), s = sinf(f);
    float q0 = q[t], q1 = q[t + 16];
    q[t]      = (q0 * c - q1 * s) * QS;
    q[t + 16] = (q1 * c + q0 * s) * QS;
    float k0 = k[t], k1 = k[t + 16];
    k[t]      = k0 * c - k1 * s;
    k[t + 16] = k1 * c + k0 * s;
  } else {
    int d = t + 16;                    // pass-through dims 32..63 (q scale only)
    q[d]      *= QS;
    q[d + 16] *= QS;
  }
}

// ---------------- 7x7 neighborhood attention, online softmax ----------------
// one wave per (pixel, head); lane owns 2 head dims; bf16 output for out-GEMM
__global__ void __launch_bounds__(256)
natten_kernel(const float* __restrict__ qkv, unsigned short* __restrict__ o) {
  int gw   = blockIdx.x * 8 + (threadIdx.x >> 5);   // global wave id
  int lane = threadIdx.x & 31;
  int head = gw & (NHEADS - 1);
  int p    = gw >> 2;
  int b = p >> 12, h = (p >> 6) & 63, w = p & 63;

  const float* qp = qkv + (size_t)p * (3 * DM) + head * DH + 2 * lane;
  float qx = qp[0], qy = qp[1];

  int h0 = min(max(h - KW / 2, 0), HH - KW);
  int w0 = min(max(w - KW / 2, 0), WW - KW);

  float m = -INFINITY, s = 0.f, ax = 0.f, ay = 0.f;
  for (int a = 0; a < KW; ++a) {
    for (int c = 0; c < KW; ++c) {
      size_t np = ((size_t)b * HH + (h0 + a)) * WW + (w0 + c);
      const float* base = qkv + np * (3 * DM) + head * DH + 2 * lane;
      const float* kp = base + DM;       // k
      float part = qx * kp[0] + qy * kp[1];
      #pragma unroll
      for (int off = 16; off; off >>= 1) part += __shfl_xor(part, off, 32);
      float mn   = fmaxf(m, part);
      float corr = __expf(m - mn);       // first iter: exp(-inf)=0
      float e    = __expf(part - mn);
      const float* vp = base + 2 * DM;   // v
      s  = s * corr + e;
      ax = ax * corr + e * vp[0];
      ay = ay * corr + e * vp[1];
      m  = mn;
    }
  }
  float inv = 1.0f / s;
  unsigned short* op = o + (size_t)p * DM + head * DH + 2 * lane;
  op[0] = f2bf(ax * inv);
  op[1] = f2bf(ay * inv);
}

// ---------------- GEGLU: g = val * gelu_exact(gate) ----------------
__global__ void geglu_kernel(const unsigned short* __restrict__ u,
                             unsigned short* __restrict__ g, int total) {
  int i = blockIdx.x * blockDim.x + threadIdx.x;
  if (i >= total) return;
  int r = i / DFF, j = i % DFF;
  float val  = bf2f(u[(size_t)r * (2 * DFF) + j]);
  float gate = bf2f(u[(size_t)r * (2 * DFF) + DFF + j]);
  float ge = 0.5f * gate * (1.0f + erff(gate * 0.70710678118f));
  g[i] = f2bf(val * ge);
}

// ---------------- host side ----------------
extern "C" void kernel_launch(void* const* d_in, const int* in_sizes, int n_in,
                              void* d_out, int out_size, void* d_ws, size_t ws_size,
                              hipStream_t stream) {
  (void)in_sizes; (void)n_in; (void)out_size; (void)ws_size;
  const float* x      = (const float*)d_in[0];
  const float* pos    = (const float*)d_in[1];
  const float* cond   = (const float*)d_in[2];
  const float* anw    = (const float*)d_in[3];   // attn_norm_w [256,256]
  const float* qkv_w  = (const float*)d_in[4];   // [768,256]
  const float* out_w  = (const float*)d_in[5];   // [256,256]
  const float* fnw    = (const float*)d_in[6];   // ff_norm_w [256,256]
  const float* up_w   = (const float*)d_in[7];   // [1536,256]
  const float* down_w = (const float*)d_in[8];   // [256,768]
  float* out = (float*)d_out;

  char* ws = (char*)d_ws;
  size_t off = 0;
  auto alloc = [&](size_t bytes) -> char* {
    char* p = ws + off;
    off += (bytes + 255) & ~(size_t)255;
    return p;
  };

  float*          attn_ada = (float*)alloc(BB * DM * sizeof(float));
  float*          ff_ada   = (float*)alloc(BB * DM * sizeof(float));
  unsigned short* qkvw_bf  = (unsigned short*)alloc((size_t)3 * DM * DM * 2);
  unsigned short* outw_bf  = (unsigned short*)alloc((size_t)DM * DM * 2);
  unsigned short* upw_bf   = (unsigned short*)alloc((size_t)2 * DFF * DM * 2);
  unsigned short* downw_bf = (unsigned short*)alloc((size_t)DM * DFF * 2);
  unsigned short* xn_bf    = (unsigned short*)alloc((size_t)NPIX * DM * 2);
  float*          qkv      = (float*)alloc((size_t)NPIX * 3 * DM * 4);   // 24 MB
  unsigned short* u_bf     = (unsigned short*)qkv;  // alias: [8192,1536] bf16 == 24 MB, qkv dead by then
  unsigned short* o_bf     = (unsigned short*)alloc((size_t)NPIX * DM * 2);
  float*          x1       = (float*)alloc((size_t)NPIX * DM * 4);
  unsigned short* xn2_bf   = (unsigned short*)alloc((size_t)NPIX * DM * 2);
  unsigned short* g_bf     = (unsigned short*)alloc((size_t)NPIX * DFF * 2);

  // 1) ada scales from cond
  ada_scale_kernel<<<dim3((BB * DM + 255) / 256), dim3(256), 0, stream>>>(cond, anw, attn_ada);
  ada_scale_kernel<<<dim3((BB * DM + 255) / 256), dim3(256), 0, stream>>>(cond, fnw, ff_ada);

  // 2) weights -> bf16
  int n;
  n = 3 * DM * DM;  cast_bf16_kernel<<<dim3((n + 255) / 256), dim3(256), 0, stream>>>(qkv_w,  qkvw_bf,  n);
  n = DM * DM;      cast_bf16_kernel<<<dim3((n + 255) / 256), dim3(256), 0, stream>>>(out_w,  outw_bf,  n);
  n = 2 * DFF * DM; cast_bf16_kernel<<<dim3((n + 255) / 256), dim3(256), 0, stream>>>(up_w,   upw_bf,   n);
  n = DM * DFF;     cast_bf16_kernel<<<dim3((n + 255) / 256), dim3(256), 0, stream>>>(down_w, downw_bf, n);

  // 3) adaRMS -> bf16
  rms_to_bf16_kernel<<<dim3(NPIX / 8), dim3(256), 0, stream>>>(x, attn_ada, xn_bf, HH * WW);

  // 4) QKV GEMM: [8192,256] x [768,256]^T -> [8192,768] f32
  gemm_bf16_wmma<false, false><<<dim3((3 * DM) / 64, NPIX / 128), dim3(256), 0, stream>>>(
      xn_bf, qkvw_bf, nullptr, qkv, NPIX, 3 * DM, DM);

  // 5) axial RoPE + q-scale in place
  rope_kernel<<<dim3((NPIX * NHEADS * 32) / 256), dim3(256), 0, stream>>>(qkv, pos);

  // 6) 7x7 neighborhood attention -> o bf16
  natten_kernel<<<dim3((NPIX * NHEADS) / 8), dim3(256), 0, stream>>>(qkv, o_bf);

  // 7) out GEMM + skip(x) -> x1 f32
  gemm_bf16_wmma<true, false><<<dim3(DM / 64, NPIX / 128), dim3(256), 0, stream>>>(
      o_bf, outw_bf, x, x1, NPIX, DM, DM);

  // 8) adaRMS (ff) -> bf16
  rms_to_bf16_kernel<<<dim3(NPIX / 8), dim3(256), 0, stream>>>(x1, ff_ada, xn2_bf, HH * WW);

  // 9) up GEMM -> u bf16 [8192,1536] (aliases qkv buffer)
  gemm_bf16_wmma<false, true><<<dim3((2 * DFF) / 64, NPIX / 128), dim3(256), 0, stream>>>(
      xn2_bf, upw_bf, nullptr, u_bf, NPIX, 2 * DFF, DM);

  // 10) GEGLU -> g bf16 [8192,768]
  geglu_kernel<<<dim3((NPIX * DFF + 255) / 256), dim3(256), 0, stream>>>(u_bf, g_bf, NPIX * DFF);

  // 11) down GEMM + skip(x1) -> out f32
  gemm_bf16_wmma<true, false><<<dim3(DM / 64, NPIX / 128), dim3(256), 0, stream>>>(
      g_bf, downw_bf, x1, out, NPIX, DM, DFF);
}
